// Model_71451075936458
// MI455X (gfx1250) — matrix operand order repaired and checked by reading, not compiled, not run on hardware
//
#include <hip/hip_runtime.h>
#include <hip/hip_bf16.h>

// ---------------------------------------------------------------------------
// Autoregressive gumbel-softmax TCN, restructured as an incremental causal
// dilated TCN over absolute time:
//   u_s = [x_{s-15} (8ch), z_s (8ch)]
//   h0[s]   = elu(W0·[u_{s-1};u_s]       + b0)           (pre0x part precomputed)
//   h1[s]   = elu(W1·[h0_{s-2};h0_s]     + b1)
//   h2[s]   = elu(W2·[h1_{s-4};h1_s]     + b2)
//   logit[s]=     W3·[h2_{s-8};h2_s]     + b3
//   s = 15..510 emit logq, y=gumbel-softmax, z_{s+1}=y  (z_{0..15}=0)
// Batch trajectories are independent -> 8 independent single-wave blocks,
// each owning 16 batch columns (WMMA N=16), no barriers, LDS ring buffers.
// ---------------------------------------------------------------------------

typedef __attribute__((ext_vector_type(16))) _Float16 v16h;
typedef __attribute__((ext_vector_type(8)))  float    v8f;

#define WMMA16(A, B, C) \
    __builtin_amdgcn_wmma_f32_16x16x32_f16(false, (A), false, (B), (short)0, (C), false, false)

__device__ __forceinline__ int kmapA(int e, int half) {
    // A-fragment (16x32 f16) per-lane K index: lanes0-15: K=e(0..15) ;
    // lanes16-31: K=16..31
    return e + (half << 4);
}
__device__ __forceinline__ float elu1(float v) { return v > 0.f ? v : (__expf(v) - 1.f); }

// ---------------------------------------------------------------------------
// Kernel 1: parallel precompute  pre0x[s][b][c] = b0[c] + W0x(:,:,0)·x_{s-16}
//                                               + W0x(:,:,1)·x_{s-15}
// via WMMA: A = W0x padded 32x32 (K>=16 zero), B = x window (rows>=16 zero).
// 4096 wave-tasks (128 b x 32 s-tiles), 8 waves / block.
// ---------------------------------------------------------------------------
__global__ __launch_bounds__(256) void tcn_pre0x_kernel(
    const float* __restrict__ x, const float* __restrict__ w0,
    const float* __restrict__ b0, float* __restrict__ pre)
{
    const int tid  = threadIdx.x;
    const int lane = tid & 31, half = lane >> 4, col = lane & 15;
    const int task = blockIdx.x * 8 + (tid >> 5);
    const int b    = task >> 5;
    const int s    = ((task & 31) << 4) + col;

    v16h AX[2];
    #pragma unroll
    for (int m = 0; m < 2; ++m) {
        #pragma unroll
        for (int e = 0; e < 16; ++e) {
            const int k = kmapA(e, half);
            float wv = 0.f;
            if (k < 8)       wv = w0[(16 * m + col) * 32 + k * 2 + 0];        // tap0, x ch k
            else if (k < 16) wv = w0[(16 * m + col) * 32 + (k - 8) * 2 + 1];  // tap1
            AX[m][e] = (_Float16)wv;
        }
    }
    v16h bx;
    #pragma unroll
    for (int e = 0; e < 16; ++e) bx[e] = (_Float16)0.f;
    if (half == 0) {  // lanes 0-15 hold K=0..15 of B; K>=16 stays zero (padding)
        const int t0 = s - 16, t1 = s - 15;
        #pragma unroll
        for (int e = 0; e < 8; ++e)
            bx[e] = (t0 >= 0) ? (_Float16)x[b * 4096 + e * 512 + t0] : (_Float16)0.f;
        #pragma unroll
        for (int e = 0; e < 8; ++e)
            bx[8 + e] = (t1 >= 0) ? (_Float16)x[b * 4096 + e * 512 + t1] : (_Float16)0.f;
    }
    #pragma unroll
    for (int m = 0; m < 2; ++m) {
        v8f c;
        #pragma unroll
        for (int i = 0; i < 8; ++i) c[i] = b0[16 * m + 8 * half + i];
        v8f d = WMMA16(AX[m], bx, c);
        float* o = pre + ((size_t)s * 128 + b) * 32 + 16 * m + 8 * half;
        float4 lo = {d[0], d[1], d[2], d[3]}, hi = {d[4], d[5], d[6], d[7]};
        *(float4*)(o) = lo;
        *(float4*)(o + 4) = hi;
    }
}

// ---------------------------------------------------------------------------
// Kernel 2: zero z_all[:, :, 0:16] (capacity prefix; d_out is poisoned)
// ---------------------------------------------------------------------------
__global__ void tcn_zinit_kernel(float* __restrict__ z)
{
    const int i = blockIdx.x * 256 + threadIdx.x;
    if (i < 128 * 8 * 16) {
        const int b = i >> 7, r = i & 127, k = r >> 4, t = r & 15;
        z[b * 4096 + k * 512 + t] = 0.f;
    }
}

// ---------------------------------------------------------------------------
// Kernel 3: serial AR loop. 8 blocks x 1 wave; block w owns batch 16w..16w+15.
// LDS rings (f16, [col][32ch] per slot): h0 x4, h1 x8, h2 x16, z x4 (ring size
// covers dilation look-back). Weights live in A-fragment registers.
// ---------------------------------------------------------------------------
__global__ __launch_bounds__(32) void tcn_ar_kernel(
    const float* __restrict__ temp, const float* __restrict__ u,
    const float* __restrict__ w0,
    const float* __restrict__ w1, const float* __restrict__ b1,
    const float* __restrict__ w2, const float* __restrict__ b2,
    const float* __restrict__ w3, const float* __restrict__ b3,
    const float* __restrict__ pre, float* __restrict__ zall,
    float* __restrict__ lq)
{
    extern __shared__ __align__(16) _Float16 smem[];
    const int H0 = 0, H1 = 2048, H2 = 6144, ZR = 14336, TOT = 14848; // f16 elems

    const int lane = threadIdx.x, half = lane >> 4, col = lane & 15;
    const int b = blockIdx.x * 16 + col;
    const float tinv = 1.0f / temp[0];

    // ---- constant A fragments (weights) and C fragments (biases) ----
    v16h A0[2], A1[2][2], A2[2][2], A3[2];
    #pragma unroll
    for (int m = 0; m < 2; ++m) {
        #pragma unroll
        for (int e = 0; e < 16; ++e) {
            const int k = kmapA(e, half);
            float wv = 0.f;
            if (k < 8)       wv = w0[(16 * m + col) * 32 + (8 + k) * 2 + 0];  // z ch, tap0
            else if (k < 16) wv = w0[(16 * m + col) * 32 + k * 2 + 1];        // z ch, tap1
            A0[m][e] = (_Float16)wv;
            #pragma unroll
            for (int kc = 0; kc < 2; ++kc) {
                A1[m][kc][e] = (_Float16)w1[(16 * m + col) * 64 + k * 2 + kc];
                A2[m][kc][e] = (_Float16)w2[(16 * m + col) * 64 + k * 2 + kc];
            }
        }
    }
    #pragma unroll
    for (int kc = 0; kc < 2; ++kc)
        #pragma unroll
        for (int e = 0; e < 16; ++e) {
            const int k = kmapA(e, half);
            A3[kc][e] = (col < 8) ? (_Float16)w3[col * 64 + k * 2 + kc] : (_Float16)0.f;
        }

    v8f CB1[2], CB2[2], CB3;
    #pragma unroll
    for (int m = 0; m < 2; ++m)
        #pragma unroll
        for (int i = 0; i < 8; ++i) {
            CB1[m][i] = b1[16 * m + 8 * half + i];
            CB2[m][i] = b2[16 * m + 8 * half + i];
        }
    #pragma unroll
    for (int i = 0; i < 8; ++i) CB3[i] = (half == 0) ? b3[i] : 0.f;

    for (int i = lane; i < TOT; i += 32) smem[i] = (_Float16)0.f;
    __syncthreads();  // single-wave WG: lowers to counter waits only

    for (int s = 1; s <= 510; ++s) {
        if (s < 510)
            __builtin_prefetch(pre + ((size_t)(s + 1) * 128 + b) * 32, 0, 1);
        if (s >= 14 && s < 510)
            __builtin_prefetch(u + ((size_t)(s - 14) * 128 + b) * 8, 0, 1);

        // ---- layer 0: h0[s] = elu(pre0x[s] + W0z·[z_{s-1};z_s]) ----
        v16h bz;
        #pragma unroll
        for (int e = 0; e < 16; ++e) bz[e] = (_Float16)0.f;
        if (half == 0) {
            const _Float16* zm1 = smem + ZR + ((s - 1) & 3) * 128 + col * 8;
            const _Float16* zs  = smem + ZR + (s & 3) * 128 + col * 8;
            #pragma unroll
            for (int e = 0; e < 8; ++e) { bz[e] = zm1[e]; bz[8 + e] = zs[e]; }
        }
        #pragma unroll
        for (int m = 0; m < 2; ++m) {
            const float* cp = pre + ((size_t)s * 128 + b) * 32 + 16 * m + 8 * half;
            const float4 lo = *(const float4*)cp;
            const float4 hi = *(const float4*)(cp + 4);
            v8f c = {lo.x, lo.y, lo.z, lo.w, hi.x, hi.y, hi.z, hi.w};
            v8f d = WMMA16(A0[m], bz, c);
            _Float16* o = smem + H0 + (s & 3) * 512 + col * 32 + 16 * m + 8 * half;
            #pragma unroll
            for (int i = 0; i < 8; ++i) o[i] = (_Float16)elu1(d[i]);
        }

        // ---- layer 1: h1[s] = elu(W1·[h0_{s-2};h0_s] + b1) ----
        if (s >= 3) {
            const _Float16* p0 = smem + H0 + ((s - 2) & 3) * 512 + col * 32 + 16 * half;
            const _Float16* p1 = smem + H0 + (s & 3) * 512 + col * 32 + 16 * half;
            v16h Bp, Bc;
            #pragma unroll
            for (int e = 0; e < 16; ++e) { Bp[e] = p0[e]; Bc[e] = p1[e]; }
            #pragma unroll
            for (int m = 0; m < 2; ++m) {
                v8f d = WMMA16(A1[m][0], Bp, CB1[m]);
                d     = WMMA16(A1[m][1], Bc, d);
                _Float16* o = smem + H1 + (s & 7) * 512 + col * 32 + 16 * m + 8 * half;
                #pragma unroll
                for (int i = 0; i < 8; ++i) o[i] = (_Float16)elu1(d[i]);
            }
        }

        // ---- layer 2: h2[s] = elu(W2·[h1_{s-4};h1_s] + b2) ----
        if (s >= 7) {
            const _Float16* p0 = smem + H1 + ((s - 4) & 7) * 512 + col * 32 + 16 * half;
            const _Float16* p1 = smem + H1 + (s & 7) * 512 + col * 32 + 16 * half;
            v16h Bp, Bc;
            #pragma unroll
            for (int e = 0; e < 16; ++e) { Bp[e] = p0[e]; Bc[e] = p1[e]; }
            #pragma unroll
            for (int m = 0; m < 2; ++m) {
                v8f d = WMMA16(A2[m][0], Bp, CB2[m]);
                d     = WMMA16(A2[m][1], Bc, d);
                _Float16* o = smem + H2 + (s & 15) * 512 + col * 32 + 16 * m + 8 * half;
                #pragma unroll
                for (int i = 0; i < 8; ++i) o[i] = (_Float16)elu1(d[i]);
            }
        }

        // ---- layer 3 + softmaxes + feedback ----
        if (s >= 15) {
            const _Float16* p0 = smem + H2 + ((s - 8) & 15) * 512 + col * 32 + 16 * half;
            const _Float16* p1 = smem + H2 + (s & 15) * 512 + col * 32 + 16 * half;
            v16h Bp, Bc;
            #pragma unroll
            for (int e = 0; e < 16; ++e) { Bp[e] = p0[e]; Bc[e] = p1[e]; }
            v8f d = WMMA16(A3[0], Bp, CB3);
            d     = WMMA16(A3[1], Bc, d);
            if (half == 0) {  // lanes 0-15 hold all 8 logits of their batch col
                float mx = d[0];
                #pragma unroll
                for (int i = 1; i < 8; ++i) mx = fmaxf(mx, d[i]);
                float se = 0.f;
                #pragma unroll
                for (int i = 0; i < 8; ++i) se += __expf(d[i] - mx);
                const float lse = __logf(se);
                float lgv[8];
                #pragma unroll
                for (int i = 0; i < 8; ++i) lgv[i] = d[i] - mx - lse;

                const int t = s - 15;
                const float* up = u + ((size_t)t * 128 + b) * 8;
                float a[8];
                float m2 = -3.0e38f;
                #pragma unroll
                for (int i = 0; i < 8; ++i) {
                    const float g = -__logf(-__logf(up[i] + 1e-10f) + 1e-10f);
                    a[i] = (lgv[i] + g) * tinv;
                    m2   = fmaxf(m2, a[i]);
                }
                float y[8], ss = 0.f;
                #pragma unroll
                for (int i = 0; i < 8; ++i) { y[i] = __expf(a[i] - m2); ss += y[i]; }
                const float inv = 1.0f / ss;
                _Float16* zo = smem + ZR + (((s + 1) & 3) * 128) + col * 8;
                #pragma unroll
                for (int i = 0; i < 8; ++i) {
                    const float yi = y[i] * inv;
                    zall[b * 4096 + i * 512 + (s + 1)] = yi;
                    lq[b * 3968 + i * 496 + t]         = lgv[i];
                    zo[i] = (_Float16)yi;
                }
            }
        }
    }
}

// ---------------------------------------------------------------------------
extern "C" void kernel_launch(void* const* d_in, const int* in_sizes, int n_in,
                              void* d_out, int out_size, void* d_ws, size_t ws_size,
                              hipStream_t stream)
{
    (void)in_sizes; (void)n_in; (void)out_size; (void)ws_size;
    const float* x    = (const float*)d_in[0];
    const float* temp = (const float*)d_in[1];
    const float* u    = (const float*)d_in[2];
    const float* w0   = (const float*)d_in[3];
    const float* b0   = (const float*)d_in[4];
    const float* w1   = (const float*)d_in[5];
    const float* b1   = (const float*)d_in[6];
    const float* w2   = (const float*)d_in[7];
    const float* b2   = (const float*)d_in[8];
    const float* w3   = (const float*)d_in[9];
    const float* b3   = (const float*)d_in[10];

    float* zall = (float*)d_out;           // (128, 8, 512)
    float* lq   = zall + 128 * 8 * 512;    // (128, 8, 496, 1)
    float* pre  = (float*)d_ws;            // pre0x: 512*128*32 f32 = 8 MB

    tcn_pre0x_kernel<<<512, 256, 0, stream>>>(x, w0, b0, pre);
    tcn_zinit_kernel<<<64, 256, 0, stream>>>(zall);
    tcn_ar_kernel<<<8, 32, 29696, stream>>>(temp, u, w0, w1, b1, w2, b2, w3, b3,
                                            pre, zall, lq);
}
